// HolographicQKV_5454608466458
// MI455X (gfx1250) — matrix operand order repaired and checked by reading, not compiled, not run on hardware
//
#include <hip/hip_runtime.h>

// ---------------------------------------------------------------------------
// HolographicQKV for MI455X (gfx1250, wave32)
//
// Pre:     x -> bf16 ; W{q,k,v} -> bf16 transposed Wt[n][k]   (one time)
// Phase 1: q/k/v = x @ W + b via v_wmma_f32_16x16x32_bf16,
//          64x64 block tile, 4 waves x 4 accumulators, LDS staged with
//          async global->LDS b128 copies (ASYNCcnt path).
// Phase 2: fused HRR in frequency domain, two-pass chunked causal scan.
// ---------------------------------------------------------------------------

typedef __attribute__((ext_vector_type(16))) __bf16 v16bf;
typedef __attribute__((ext_vector_type(8)))  __bf16 v8bf;
typedef __attribute__((ext_vector_type(8)))  float  v8f;
typedef __attribute__((ext_vector_type(4)))  int    v4i;

#define BB 8
#define SS 2048
#define DD 1024
#define HH 16
#define HD 64
#define MM (BB * SS)       // 16384 rows
#define CH 16              // scan chunks over S
#define CL (SS / CH)       // 128 tokens per chunk

#if __has_builtin(__builtin_amdgcn_global_load_async_to_lds_b128) && \
    __has_builtin(__builtin_amdgcn_s_wait_asynccnt)
#define USE_ASYNC_LDS 1
#else
#define USE_ASYNC_LDS 0
#endif

// ---------------------------------------------------------------------------
// Pre-pass A: elementwise f32 -> bf16 (x), 8 elements per thread.
// ---------------------------------------------------------------------------
__global__ __launch_bounds__(256)
void cvt_bf16(const float* __restrict__ src, __bf16* __restrict__ dst)
{
    const size_t i = ((size_t)blockIdx.x * 256 + threadIdx.x) * 8;
    const float4 a = *(const float4*)(src + i);
    const float4 b = *(const float4*)(src + i + 4);
    v8bf o;
    o[0] = (__bf16)a.x; o[1] = (__bf16)a.y; o[2] = (__bf16)a.z; o[3] = (__bf16)a.w;
    o[4] = (__bf16)b.x; o[5] = (__bf16)b.y; o[6] = (__bf16)b.z; o[7] = (__bf16)b.w;
    *(v8bf*)(dst + i) = o;
}

// ---------------------------------------------------------------------------
// Pre-pass B: W[k][n] (f32) -> Wt[n][k] (bf16), 32x32 LDS tile transpose.
// blockIdx.z selects which of the three W matrices.
// ---------------------------------------------------------------------------
__global__ __launch_bounds__(256)
void transpose_w_bf16(const float* __restrict__ Wq, const float* __restrict__ Wk,
                      const float* __restrict__ Wv,
                      __bf16* __restrict__ T0, __bf16* __restrict__ T1,
                      __bf16* __restrict__ T2)
{
    const int which = blockIdx.z;
    const float* __restrict__ W  = (which == 0) ? Wq : (which == 1) ? Wk : Wv;
    __bf16* __restrict__      Wt = (which == 0) ? T0 : (which == 1) ? T1 : T2;

    __shared__ float tile[32][33];
    const int tx = threadIdx.x;          // 0..31
    const int ty = threadIdx.y;          // 0..7
    const int k0 = blockIdx.x * 32;
    const int n0 = blockIdx.y * 32;

#pragma unroll
    for (int j = 0; j < 4; ++j)
        tile[ty + 8 * j][tx] = W[(size_t)(k0 + ty + 8 * j) * DD + (n0 + tx)];
    __syncthreads();
#pragma unroll
    for (int j = 0; j < 4; ++j)
        Wt[(size_t)(n0 + ty + 8 * j) * DD + (k0 + tx)] =
            (__bf16)tile[tx][ty + 8 * j];
}

// ---------------------------------------------------------------------------
// Kernel 1: bf16 WMMA GEMM. Block = 128 threads = 4 waves.
// Block tile 64(M) x 64(N); wave w owns rows [w*16, w*16+16) x all 64 cols.
// Inputs already bf16; B already transposed -> staging is pure b128 copies.
// ---------------------------------------------------------------------------
__global__ __launch_bounds__(128)
void qkv_gemm_wmma(const __bf16* __restrict__ xb,
                   const __bf16* __restrict__ Wt0, const __bf16* __restrict__ Wt1,
                   const __bf16* __restrict__ Wt2,
                   const float* __restrict__ bq, const float* __restrict__ bk,
                   const float* __restrict__ bv,
                   float* __restrict__ q, float* __restrict__ k,
                   float* __restrict__ v)
{
    const int tile_m = blockIdx.x;
    const int tile_n = blockIdx.y;
    const int which  = blockIdx.z;

    const __bf16* __restrict__ Wt  = (which == 0) ? Wt0 : (which == 1) ? Wt1 : Wt2;
    const float*  __restrict__ bias = (which == 0) ? bq : (which == 1) ? bk : bv;
    float* __restrict__        Y    = (which == 0) ? q  : (which == 1) ? k  : v;

    const int tid  = threadIdx.x;        // 0..127
    const int wave = tid >> 5;
    const int lane = tid & 31;

    // 80-byte row stride (20 banks): conflict-free b128 fragment reads.
    __shared__ __align__(16) __bf16 lds_a[64][40];   // x rows  [m][k]
    __shared__ __align__(16) __bf16 lds_b[64][40];   // Wt rows [n][k]

    const int m0 = tile_m * 64;
    const int n0 = tile_n * 64;

    v8f acc0 = {0,0,0,0,0,0,0,0};
    v8f acc1 = {0,0,0,0,0,0,0,0};
    v8f acc2 = {0,0,0,0,0,0,0,0};
    v8f acc3 = {0,0,0,0,0,0,0,0};

    for (int k0 = 0; k0 < DD; k0 += 32) {
        // Stage A and B tiles: 64 rows x 32 bf16 (64 B) each, as 16-B chunks.
        // 256 chunks per tile, 2 per thread per tile.
#pragma unroll
        for (int i = 0; i < 2; ++i) {
            const int fi  = tid + i * 128;       // 0..255
            const int r   = fi >> 2;             // row 0..63
            const int seg = (fi & 3) * 8;        // 8 bf16 = 16 B
            const __bf16* ga = xb + (size_t)(m0 + r) * DD + k0 + seg;
            const __bf16* gb = Wt + (size_t)(n0 + r) * DD + k0 + seg;
#if USE_ASYNC_LDS
            __builtin_amdgcn_global_load_async_to_lds_b128(
                (__attribute__((address_space(1))) v4i*)(size_t)ga,
                (__attribute__((address_space(3))) v4i*)&lds_a[r][seg], 0, 0);
            __builtin_amdgcn_global_load_async_to_lds_b128(
                (__attribute__((address_space(1))) v4i*)(size_t)gb,
                (__attribute__((address_space(3))) v4i*)&lds_b[r][seg], 0, 0);
#else
            *(v8bf*)&lds_a[r][seg] = *(const v8bf*)ga;
            *(v8bf*)&lds_b[r][seg] = *(const v8bf*)gb;
#endif
        }
#if USE_ASYNC_LDS
        __builtin_amdgcn_s_wait_asynccnt(0);
#endif
        __syncthreads();

        // A fragment (ISA 7.12.2, 16-bit A 16x32): lane&15 = M row,
        // lanes 0-15 hold K {0..7,16..23}, lanes 16-31 hold K {8..15,24..31}
        const int mrow = wave * 16 + (lane & 15);
        const int koff = (lane >> 4) ? 8 : 0;
        const v8bf alo = *(const v8bf*)&lds_a[mrow][koff];
        const v8bf ahi = *(const v8bf*)&lds_a[mrow][koff + 16];
        const v16bf afrag = __builtin_shufflevector(
            alo, ahi, 0, 1, 2, 3, 4, 5, 6, 7, 8, 9, 10, 11, 12, 13, 14, 15);

        // B fragments (B 32x16): lane&15 = N col; lanes 0-15 K {0..7,16..23},
        // lanes 16-31 K {8..15,24..31}; from lds_b[n][k] -> b128 loads
        const int kb = (lane >> 4) ? 8 : 0;
#pragma unroll
        for (int t = 0; t < 4; ++t) {
            const int nloc = t * 16 + (lane & 15);
            const v8bf blo = *(const v8bf*)&lds_b[nloc][kb];
            const v8bf bhi = *(const v8bf*)&lds_b[nloc][kb + 16];
            const v16bf bfrag = __builtin_shufflevector(
                blo, bhi, 0, 1, 2, 3, 4, 5, 6, 7, 8, 9, 10, 11, 12, 13, 14, 15);
            v8f* accp = (t == 0) ? &acc0 : (t == 1) ? &acc1
                                          : (t == 2) ? &acc2 : &acc3;
            *accp = __builtin_amdgcn_wmma_f32_16x16x32_bf16(
                false, afrag, false, bfrag, (short)0, *accp, false, false);
        }
        __syncthreads();
    }

    // Store C: VGPR i holds M = i + 8*(lane>=16), N = lane&15 per 16x16 tile
    const int mbase = m0 + wave * 16 + ((lane >> 4) ? 8 : 0);
#pragma unroll
    for (int t = 0; t < 4; ++t) {
        const int n  = n0 + t * 16 + (lane & 15);
        const float bb = bias[n];
        const v8f acc = (t == 0) ? acc0 : (t == 1) ? acc1
                                         : (t == 2) ? acc2 : acc3;
#pragma unroll
        for (int i = 0; i < 8; ++i)
            Y[(size_t)(mbase + i) * DD + n] = acc[i] + bb;
    }
}

// ---------------------------------------------------------------------------
// Kernel 2a: per-chunk spectral sums of bind(k,v).
// Block = (b,h,chunk), 64 threads; thread f owns frequency bin f.
// ---------------------------------------------------------------------------
__global__ __launch_bounds__(64)
void hrr_chunksum(const float* __restrict__ k, const float* __restrict__ v,
                  float* __restrict__ T)
{
    const int bhc = blockIdx.x;
    const int c   = bhc % CH;
    const int h   = (bhc / CH) % HH;
    const int b   = bhc / (CH * HH);
    const int f   = threadIdx.x;

    __shared__ float tw_c[HD], tw_s[HD];
    __shared__ float rk[HD], rv[HD];

    {
        const float ang = (2.0f * 3.14159265358979323846f / (float)HD) * (float)f;
        tw_c[f] = cosf(ang);
        tw_s[f] = sinf(ang);
    }
    __syncthreads();

    float Ar = 0.0f, Ai = 0.0f;

    for (int s = c * CL; s < (c + 1) * CL; ++s) {
        const size_t base = ((size_t)(b * SS + s)) * DD + (size_t)h * HD;
        rk[f] = k[base + f];
        rv[f] = v[base + f];
        __syncthreads();

        float Kr = 0.f, Ki = 0.f, Vr = 0.f, Vi = 0.f;
#pragma unroll 8
        for (int n = 0; n < HD; ++n) {
            const int idx = (f * n) & (HD - 1);
            const float cc = tw_c[idx];
            const float sn = tw_s[idx];
            const float xk = rk[n], xv = rv[n];
            Kr = fmaf(xk, cc, Kr);  Ki = fmaf(xk, -sn, Ki);
            Vr = fmaf(xv, cc, Vr);  Vi = fmaf(xv, -sn, Vi);
        }
        float d;
        d = 1.0f / (sqrtf(Kr * Kr + Ki * Ki) + 1e-8f); Kr *= d; Ki *= d;
        d = 1.0f / (sqrtf(Vr * Vr + Vi * Vi) + 1e-8f); Vr *= d; Vi *= d;

        Ar += Kr * Vr - Ki * Vi;
        Ai += Kr * Vi + Ki * Vr;
        __syncthreads();
    }

    const size_t ti = ((((size_t)(b * HH + h)) * CH + c) * 2) * HD;
    T[ti + f]      = Ar;
    T[ti + HD + f] = Ai;
}

// ---------------------------------------------------------------------------
// Kernel 2b: replay each chunk with exclusive prefix of chunk totals.
// ---------------------------------------------------------------------------
__global__ __launch_bounds__(64)
void hrr_scan(const float* __restrict__ q, const float* __restrict__ k,
              const float* __restrict__ v, const float* __restrict__ T,
              float* __restrict__ out)
{
    const int bhc = blockIdx.x;
    const int c   = bhc % CH;
    const int h   = (bhc / CH) % HH;
    const int b   = bhc / (CH * HH);
    const int f   = threadIdx.x;

    __shared__ float tw_c[HD], tw_s[HD];
    __shared__ float rq[HD], rk[HD], rv[HD];
    __shared__ float gr[HD], gi[HD];

    {
        const float ang = (2.0f * 3.14159265358979323846f / (float)HD) * (float)f;
        tw_c[f] = cosf(ang);
        tw_s[f] = sinf(ang);
    }

    // exclusive prefix of chunk totals for this (b,h), bin f
    float Ar = 0.0f, Ai = 0.0f;
    for (int cc = 0; cc < c; ++cc) {
        const size_t ti = ((((size_t)(b * HH + h)) * CH + cc) * 2) * HD;
        Ar += T[ti + f];
        Ai += T[ti + HD + f];
    }
    __syncthreads();

    for (int s = c * CL; s < (c + 1) * CL; ++s) {
        const size_t base = ((size_t)(b * SS + s)) * DD + (size_t)h * HD;
        rq[f] = q[base + f];
        rk[f] = k[base + f];
        rv[f] = v[base + f];
        __syncthreads();

        float Qr = 0.f, Qi = 0.f, Kr = 0.f, Ki = 0.f, Vr = 0.f, Vi = 0.f;
#pragma unroll 8
        for (int n = 0; n < HD; ++n) {
            const int idx = (f * n) & (HD - 1);
            const float cc2 = tw_c[idx];
            const float sn  = tw_s[idx];
            const float xq = rq[n], xk = rk[n], xv = rv[n];
            Qr = fmaf(xq, cc2, Qr);  Qi = fmaf(xq, -sn, Qi);
            Kr = fmaf(xk, cc2, Kr);  Ki = fmaf(xk, -sn, Ki);
            Vr = fmaf(xv, cc2, Vr);  Vi = fmaf(xv, -sn, Vi);
        }
        float d;
        d = 1.0f / (sqrtf(Qr * Qr + Qi * Qi) + 1e-8f); Qr *= d; Qi *= d;
        d = 1.0f / (sqrtf(Kr * Kr + Ki * Ki) + 1e-8f); Kr *= d; Ki *= d;
        d = 1.0f / (sqrtf(Vr * Vr + Vi * Vi) + 1e-8f); Vr *= d; Vi *= d;

        Ar += Kr * Vr - Ki * Vi;
        Ai += Kr * Vi + Ki * Vr;

        const float Gr = Ar * Qr + Ai * Qi;
        const float Gi = Ai * Qr - Ar * Qi;

        gr[f] = Gr;
        gi[f] = Gi;
        __syncthreads();

        // IDFT real part: (1/64) * sum_m (Gr*cos - Gi*sin)(2*pi*m*f/64)
        float o = 0.0f;
#pragma unroll 8
        for (int m = 0; m < HD; ++m) {
            const int idx = (f * m) & (HD - 1);
            o = fmaf(gr[m], tw_c[idx], o);
            o = fmaf(-gi[m], tw_s[idx], o);
        }
        out[base + f] = o * (1.0f / (float)HD);
    }
}

// ---------------------------------------------------------------------------
// Launch
// ---------------------------------------------------------------------------
extern "C" void kernel_launch(void* const* d_in, const int* in_sizes, int n_in,
                              void* d_out, int out_size, void* d_ws,
                              size_t ws_size, hipStream_t stream)
{
    (void)in_sizes; (void)n_in; (void)out_size; (void)ws_size;

    const float* x  = (const float*)d_in[0];
    const float* Wq = (const float*)d_in[1];
    const float* bq = (const float*)d_in[2];
    const float* Wk = (const float*)d_in[3];
    const float* bk = (const float*)d_in[4];
    const float* Wv = (const float*)d_in[5];
    const float* bv = (const float*)d_in[6];
    float* out = (float*)d_out;

    // workspace: q | k | v (64 MB each) | T (1 MB) | xb (32MB) | Wt x3 (2MB ea)
    float*  q   = (float*)d_ws;
    float*  k   = q + (size_t)MM * DD;
    float*  v   = k + (size_t)MM * DD;
    float*  T   = v + (size_t)MM * DD;                       // [BB][HH][CH][2][HD]
    __bf16* xb  = (__bf16*)(T + (size_t)BB * HH * CH * 2 * HD);
    __bf16* Wt0 = xb + (size_t)MM * DD;
    __bf16* Wt1 = Wt0 + (size_t)DD * DD;
    __bf16* Wt2 = Wt1 + (size_t)DD * DD;

    // pre-pass: convert/transpose
    cvt_bf16<<<dim3((MM * (size_t)DD) / (256 * 8)), 256, 0, stream>>>(x, xb);
    transpose_w_bf16<<<dim3(DD / 32, DD / 32, 3), dim3(32, 8), 0, stream>>>(
        Wq, Wk, Wv, Wt0, Wt1, Wt2);

    dim3 g1(MM / 64, DD / 64, 3);     // (256, 16, 3)
    qkv_gemm_wmma<<<g1, 128, 0, stream>>>(xb, Wt0, Wt1, Wt2, bq, bk, bv, q, k, v);

    hrr_chunksum<<<dim3(BB * HH * CH), 64, 0, stream>>>(k, v, T);
    hrr_scan<<<dim3(BB * HH * CH), 64, 0, stream>>>(q, k, v, T, out);
}
